// DIFFormerConv_55336358642790
// MI455X (gfx1250) — matrix operand order repaired and checked by reading, not compiled, not run on hardware
//
#include <hip/hip_runtime.h>
#include <hip/hip_bf16.h>

typedef __attribute__((ext_vector_type(16))) _Float16 v16h;
typedef __attribute__((ext_vector_type(8)))  _Float16 v8h;
typedef __attribute__((ext_vector_type(8)))  float    v8f;

#define IN_C    128
#define OUT_C   32
#define N_HEADS 4
#define TM      64          // rows per GEMM block
#define WSTR    136         // LDS row pitch in halves (272B: 16B-aligned, 68-dword bank stride)

__device__ __forceinline__ float waveSum32(float v) {
    #pragma unroll
    for (int m = 16; m >= 1; m >>= 1) v += __shfl_xor(v, m, 32);
    return v;
}

// ---------------------------------------------------------------- init
__global__ void k_zero(float* p, int n) {
    int i = blockIdx.x * blockDim.x + threadIdx.x;
    if (i < n) p[i] = 0.0f;
}

// ---------------------------------------------------------------- weight prep: f32 [k][n] -> f16 transposed [mat][n][k]
__global__ void k_prep_w(const float* __restrict__ Wq, const float* __restrict__ Wk,
                         const float* __restrict__ Wv, _Float16* __restrict__ wt) {
    int i = blockIdx.x * blockDim.x + threadIdx.x;
    if (i >= 3 * IN_C * IN_C) return;
    int mat = i >> 14, rem = i & 16383;
    int n = rem >> 7, kk = rem & 127;
    const float* W = (mat == 0) ? Wq : (mat == 1) ? Wk : Wv;
    wt[i] = (_Float16)W[kk * IN_C + n];
}

// ---------------------------------------------------------------- fused QKV GEMM (WMMA f16 -> f32)
// Weights arrive via async global->LDS DMA; X tile converted in VALU concurrently.
__global__ void __launch_bounds__(256)
k_gemm_qkv(const float* __restrict__ Xq, const float* __restrict__ Xs,
           const _Float16* __restrict__ wt,
           const float* __restrict__ bq, const float* __restrict__ bk,
           const float* __restrict__ bv,
           float* __restrict__ qO, float* __restrict__ kO, float* __restrict__ vO,
           int N) {
    __shared__ _Float16 sW[3 * IN_C * WSTR];   // transposed: [mat][n][k], padded rows
    __shared__ _Float16 sX2[2 * TM * WSTR];    // [src][row][k]

    const int tid = threadIdx.x;
    const int rowBase = blockIdx.x * TM;

    // LDS byte offset of sW (address-space(3) cast strips the flat aperture)
    unsigned swBase = (unsigned)(unsigned long long)
        (__attribute__((address_space(3))) const _Float16*)sW;

    // 1) issue async weight DMA: 3 mats * 128 rows * 16 chunks of 16B = 6144 chunks
    {
        unsigned long long src0 = (unsigned long long)(uintptr_t)wt;
        for (int c = tid; c < 3 * IN_C * 16; c += 256) {
            int mat = c >> 11;
            int rc  = c & 2047;                 // row*16 + sub within matrix
            unsigned dst = swBase + (unsigned)mat * (IN_C * WSTR * 2)
                         + (unsigned)(rc >> 4) * (WSTR * 2)
                         + (unsigned)(rc & 15) * 16u;
            unsigned long long src = src0 + (unsigned long long)c * 16ull;
            asm volatile("global_load_async_to_lds_b128 %0, %1, off"
                         :: "v"(dst), "v"(src) : "memory");
        }
    }

    // 2) meanwhile: stage X tiles f32->f16 (zero-pad tail rows)
    _Float16* sXq = sX2;
    _Float16* sXs = sX2 + TM * WSTR;
    for (int idx = tid; idx < TM * IN_C; idx += 256) {
        int r = idx >> 7, kk = idx & 127;
        int g = rowBase + r;
        float xq = (g < N) ? Xq[(size_t)g * IN_C + kk] : 0.0f;
        float xs = (g < N) ? Xs[(size_t)g * IN_C + kk] : 0.0f;
        sXq[r * WSTR + kk] = (_Float16)xq;
        sXs[r * WSTR + kk] = (_Float16)xs;
    }

    asm volatile("s_wait_asynccnt 0" ::: "memory");
    __syncthreads();

    const int wv   = tid >> 5;
    const int lane = tid & 31;
    const int l15  = lane & 15;
    const int kA   = (lane & 16) ? 8 : 0;     // A: 16-bit layout, K interleave by 8
    const int kB   = (lane & 16) ? 8 : 0;     // B: 16-bit layout, K interleave by 8
    const bool fullTile = (rowBase + TM) <= N;

    // 3 matrices x 4 M-tiles x 8 N-tiles = 96 jobs over 8 waves
    for (int j = wv; j < 96; j += 8) {
        int mat = j >> 5;
        int rem = j & 31;
        int mtile = rem >> 3;
        int ntile = rem & 7;
        const _Float16* sX = (mat == 0) ? sXq : sXs;
        const _Float16* sWm = sW + mat * (IN_C * WSTR);
        const float* bias  = (mat == 0) ? bq  : (mat == 1) ? bk  : bv;
        float* outp        = (mat == 0) ? qO  : (mat == 1) ? kO  : vO;

        const int mrow = mtile * 16 + l15;  // local A row for this lane
        const int ncol = ntile * 16 + l15;  // output column for this lane
        v8f c = {};
        #pragma unroll
        for (int kb = 0; kb < IN_C; kb += 32) {
            const _Float16* ap = sX + mrow * WSTR + kb + kA;
            v8h alo = *(const v8h*)(ap);
            v8h ahi = *(const v8h*)(ap + 16);
            v16h a = __builtin_shufflevector(alo, ahi,
                        0,1,2,3,4,5,6,7,8,9,10,11,12,13,14,15);
            const _Float16* bp = sWm + ncol * WSTR + kb + kB;
            v8h blo = *(const v8h*)(bp);
            v8h bhi = *(const v8h*)(bp + 16);
            v16h b = __builtin_shufflevector(blo, bhi,
                        0,1,2,3,4,5,6,7,8,9,10,11,12,13,14,15);
            c = __builtin_amdgcn_wmma_f32_16x16x32_f16(
                    false, a, false, b, (short)0, c, false, false);
        }
        float bb = bias[ncol];
        int rowOff = rowBase + mtile * 16 + ((lane & 16) ? 8 : 0);
        if (fullTile) {
            #pragma unroll
            for (int r = 0; r < 8; ++r)
                outp[(size_t)(rowOff + r) * IN_C + ncol] = c[r] + bb;
        } else {
            #pragma unroll
            for (int r = 0; r < 8; ++r) {
                int gr = rowOff + r;
                if (gr < N) outp[(size_t)gr * IN_C + ncol] = c[r] + bb;
            }
        }
    }
}

// ---------------------------------------------------------------- per-node: normalize q in place, vbar, vs_sum
__global__ void __launch_bounds__(256)
k_qv_node(float* __restrict__ q, const float* __restrict__ v,
          float* __restrict__ vbar, float* __restrict__ vs_sum, int N) {
    const int lane = threadIdx.x & 31;
    const int gw = blockIdx.x * 8 + (threadIdx.x >> 5);
    const int GW = gridDim.x * 8;
    float vs0 = 0, vs1 = 0, vs2 = 0, vs3 = 0;
    for (int n = gw; n < N; n += GW) {
        size_t base = (size_t)n * IN_C + lane;
        float q0 = q[base], q1 = q[base + 32], q2 = q[base + 64], q3 = q[base + 96];
        float s0 = waveSum32(q0 * q0), s1 = waveSum32(q1 * q1);
        float s2 = waveSum32(q2 * q2), s3 = waveSum32(q3 * q3);
        q[base]      = q0 * __frsqrt_rn(fmaxf(s0, 1e-30f));
        q[base + 32] = q1 * __frsqrt_rn(fmaxf(s1, 1e-30f));
        q[base + 64] = q2 * __frsqrt_rn(fmaxf(s2, 1e-30f));
        q[base + 96] = q3 * __frsqrt_rn(fmaxf(s3, 1e-30f));
        float v0 = v[base], v1 = v[base + 32], v2 = v[base + 64], v3 = v[base + 96];
        vbar[(size_t)n * OUT_C + lane] = 0.25f * (v0 + v1 + v2 + v3);
        vs0 += v0; vs1 += v1; vs2 += v2; vs3 += v3;
    }
    unsafeAtomicAdd(&vs_sum[lane],      vs0);
    unsafeAtomicAdd(&vs_sum[lane + 32], vs1);
    unsafeAtomicAdd(&vs_sum[lane + 64], vs2);
    unsafeAtomicAdd(&vs_sum[lane + 96], vs3);
}

// ---------------------------------------------------------------- kvs + ks_sum (one head per blockIdx.y)
__global__ void __launch_bounds__(256)
k_kvs(const float* __restrict__ k, const float* __restrict__ v,
      float* __restrict__ kvs, float* __restrict__ ks_sum, int N) {
    const int h = blockIdx.y;
    const int lane = threadIdx.x & 31;         // lane = m index (kvs row)
    const int gw = blockIdx.x * 8 + (threadIdx.x >> 5);
    const int GW = gridDim.x * 8;
    float acc[32];
    #pragma unroll
    for (int d = 0; d < 32; ++d) acc[d] = 0.0f;
    float ksacc = 0.0f;
    for (int n = gw; n < N; n += GW) {
        size_t base = (size_t)n * IN_C + h * OUT_C + lane;
        float km = k[base];
        float s = waveSum32(km * km);
        float kn = km * __frsqrt_rn(fmaxf(s, 1e-30f));
        ksacc += kn;
        float vd = v[base];
        #pragma unroll
        for (int d = 0; d < 32; ++d)
            acc[d] += kn * __shfl(vd, d, 32);   // v broadcast from lane d
    }
    #pragma unroll
    for (int d = 0; d < 32; ++d)
        unsafeAtomicAdd(&kvs[((h * 32 + lane) * 32) + d], acc[d]);
    unsafeAtomicAdd(&ks_sum[h * 32 + lane], ksacc);
}

// ---------------------------------------------------------------- degree histogram
__global__ void k_degree(const int* __restrict__ col, float* __restrict__ deg, int E) {
    int e = blockIdx.x * blockDim.x + threadIdx.x;
    if (e < E) unsafeAtomicAdd(&deg[col[e]], 1.0f);
}

// ---------------------------------------------------------------- attention readout (writes out)
__global__ void __launch_bounds__(256)
k_attn(const float* __restrict__ q, const float* __restrict__ kvs,
       const float* __restrict__ ks_sum, const float* __restrict__ vs_sum,
       float* __restrict__ out, int N) {
    __shared__ float skvs[N_HEADS * 32 * 32];
    __shared__ float sks[N_HEADS * 32];
    __shared__ float svs[N_HEADS * 32];
    for (int i = threadIdx.x; i < N_HEADS * 32 * 32; i += 256) skvs[i] = kvs[i];
    for (int i = threadIdx.x; i < N_HEADS * 32; i += 256) { sks[i] = ks_sum[i]; svs[i] = vs_sum[i]; }
    __syncthreads();
    const int lane = threadIdx.x & 31;         // lane = d (output channel)
    const int gw = blockIdx.x * 8 + (threadIdx.x >> 5);
    const int GW = gridDim.x * 8;
    const float fN = (float)N;
    for (int n = gw; n < N; n += GW) {
        float o = 0.0f;
        #pragma unroll
        for (int h = 0; h < N_HEADS; ++h) {
            float qm = q[(size_t)n * IN_C + h * 32 + lane];     // lane = m here
            float nrm = waveSum32(qm * sks[h * 32 + lane]) + fN;
            float a = svs[h * 32 + lane];
            #pragma unroll
            for (int m = 0; m < 32; ++m)
                a += __shfl(qm, m, 32) * skvs[(h * 32 + m) * 32 + lane];
            o += a / nrm;
        }
        out[(size_t)n * OUT_C + lane] = 0.25f * o;
    }
}

// ---------------------------------------------------------------- GCN scatter (adds into out), wave per edge
__global__ void __launch_bounds__(256)
k_scatter(const int* __restrict__ ei, const float* __restrict__ deg,
          const float* __restrict__ vbar, float* __restrict__ out, int E) {
    const int lane = threadIdx.x & 31;
    const int gw = blockIdx.x * 8 + (threadIdx.x >> 5);
    const int GW = gridDim.x * 8;
    for (int e = gw; e < E; e += GW) {
        int row = ei[e];
        int col = ei[E + e];
        float dr = deg[row], dc = deg[col];
        float val = (dr > 0.0f && dc > 0.0f) ? __frsqrt_rn(dr * dc) : 0.0f;
        float m = val * vbar[(size_t)row * OUT_C + lane];
        unsafeAtomicAdd(&out[(size_t)col * OUT_C + lane], m);
    }
}

// ================================================================ launch
extern "C" void kernel_launch(void* const* d_in, const int* in_sizes, int n_in,
                              void* d_out, int out_size, void* d_ws, size_t ws_size,
                              hipStream_t stream) {
    const float* Xq = (const float*)d_in[0];
    const float* Xs = (const float*)d_in[1];
    const float* Wq = (const float*)d_in[2];
    const float* bq = (const float*)d_in[3];
    const float* Wk = (const float*)d_in[4];
    const float* bk = (const float*)d_in[5];
    const float* Wv = (const float*)d_in[6];
    const float* bv = (const float*)d_in[7];
    const int*   ei = (const int*)d_in[8];

    const int N = in_sizes[0] / IN_C;
    const int E = in_sizes[8] / 2;

    // workspace layout (floats)
    float* ws = (float*)d_ws;
    size_t off = 0;
    float* q      = ws + off; off += (size_t)N * IN_C;
    float* k      = ws + off; off += (size_t)N * IN_C;
    float* v      = ws + off; off += (size_t)N * IN_C;
    float* vbar   = ws + off; off += (size_t)N * OUT_C;
    float* deg    = ws + off; off += (size_t)N;
    float* kvs    = ws + off; off += N_HEADS * 32 * 32;
    float* ks_sum = ws + off; off += N_HEADS * 32;
    float* vs_sum = ws + off; off += N_HEADS * 32;
    _Float16* wt  = (_Float16*)(ws + off); off += (3 * IN_C * IN_C) / 2;
    float* out    = (float*)d_out;

    // 0) zero accumulators (deg, kvs, ks_sum, vs_sum are contiguous)
    int nz = N + N_HEADS * 32 * 32 + 2 * N_HEADS * 32;
    k_zero<<<(nz + 255) / 256, 256, 0, stream>>>(deg, nz);

    // 0b) convert + transpose weights to f16 once
    k_prep_w<<<(3 * IN_C * IN_C + 255) / 256, 256, 0, stream>>>(Wq, Wk, Wv, wt);

    // 1) q,k,v = X@W + b via WMMA (weights via async global->LDS DMA)
    k_gemm_qkv<<<(N + TM - 1) / TM, 256, 0, stream>>>(
        Xq, Xs, wt, bq, bk, bv, q, k, v, N);

    // 2) normalize q in place; vbar = mean_h v; vs_sum
    k_qv_node<<<240, 256, 0, stream>>>(q, v, vbar, vs_sum, N);

    // 3) kvs = sum_n kn^T v ; ks_sum
    dim3 gkvs(48, N_HEADS, 1);
    k_kvs<<<gkvs, 256, 0, stream>>>(k, v, kvs, ks_sum, N);

    // 4) in-degree
    k_degree<<<(E + 255) / 256, 256, 0, stream>>>(ei + E, deg, E);

    // 5) attention readout -> out (overwrite)
    k_attn<<<240, 256, 0, stream>>>(q, kvs, ks_sum, vs_sum, out, N);

    // 6) GCN scatter of vbar -> out (atomic add)
    k_scatter<<<4096, 256, 0, stream>>>(ei, deg, vbar, out, E);
}
